// EyeXpertMModel_42923903156964
// MI455X (gfx1250) — compile-verified, matrix operand order checked by reading, not run on hardware
//
#include <hip/hip_runtime.h>

// ---------------------------------------------------------------------------
// EyeXpertM pipeline for MI455X (gfx1250, wave32, WMMA).
// Round 3: GEMM B-tiles staged through LDS via global_load_async_to_lds_b128
// (ASYNCcnt path, double-buffered), B fragments via ds_load_b128, A fragments
// software-pipelined in registers -> WMMAs no longer serialize on
// s_wait_loadcnt 0, and the 8x redundant global B traffic is gone.
// ---------------------------------------------------------------------------

#define B_   32
#define L_   1024
#define NW_  512
#define H_   768
#define HID_ 384
#define E_   4
#define DL_  128
#define G4_  1536   // 4*HID

#define BROW_ 40            // padded LDS row stride (elems): 80B, 16B-aligned
#define BTILE_ (128 * BROW_)

typedef unsigned short u16;
typedef __attribute__((ext_vector_type(16))) __bf16        v16bf;
typedef __attribute__((ext_vector_type(8)))  float         v8f;
typedef __attribute__((ext_vector_type(4)))  unsigned int  v4u;

__device__ __forceinline__ u16 f2bf(float f) {
  union { float f; unsigned u; } v; v.f = f;
  unsigned r = v.u + 0x7FFF + ((v.u >> 16) & 1);   // round-to-nearest-even
  return (u16)(r >> 16);
}
__device__ __forceinline__ float bf2f(u16 h) {
  union { unsigned u; float f; } v; v.u = ((unsigned)h) << 16;
  return v.f;
}
__device__ __forceinline__ float sigmoidf_(float x) {
  return 1.0f / (1.0f + __expf(-x));
}

union Frag16 { v4u q[2]; v16bf v; };

// Load a 16x32 bf16 WMMA A/B fragment from a row-major [rows, ld] bf16 array.
// ISA layout (CDNA5 ISA): lane<16 -> K = {k0..k0+7} in v0-3, {k0+16..k0+23}
// in v4-7; lane>=16 shifted by +8. Two 16B loads per lane.
__device__ __forceinline__ v16bf load_frag_ab(const u16* p, int ld, int row0, int k0) {
  int lane = threadIdx.x & 31;
  int r  = row0 + (lane & 15);
  int kk = k0 + ((lane >> 4) << 3);
  const u16* base = p + r * ld + kk;
  Frag16 f;
  f.q[0] = *(const v4u*)(base);        // K = kk .. kk+7
  f.q[1] = *(const v4u*)(base + 16);   // K = kk+16 .. kk+23
  return f.v;
}

// Async copy of one 16B chunk global -> LDS (GLOBAL_LOAD_ASYNC_TO_LDS_B128,
// tracked with ASYNCcnt; no VGPR staging, overlaps with WMMA issue).
__device__ __forceinline__ void async_g2l_b128(unsigned lds_addr, const void* gaddr) {
  asm volatile("global_load_async_to_lds_b128 %0, %1, off"
               :: "v"(lds_addr), "v"(gaddr) : "memory");
}

// ---------------------------------------------------------------------------
// 1) fp32 -> bf16 weight conversion, bias folding
// ---------------------------------------------------------------------------
__global__ void cvt_kernel(const float* __restrict__ src, u16* __restrict__ dst, int n) {
  int i = blockIdx.x * blockDim.x + threadIdx.x;
  if (i < n) dst[i] = f2bf(src[i]);
}

__global__ void bias_sum_kernel(const float* __restrict__ a, const float* __restrict__ b,
                                float* __restrict__ o, int n) {
  int i = blockIdx.x * blockDim.x + threadIdx.x;
  if (i < n) o[i] = a[i] + b[i];
}

// ---------------------------------------------------------------------------
// 2) Segment-mean pooling
// ---------------------------------------------------------------------------
__global__ void seg_mean_kernel(const float* __restrict__ tok, const int* __restrict__ wmap,
                                u16* __restrict__ wemb) {
  int w = blockIdx.x, b = blockIdx.y;
  const int* m = wmap + b * L_;
  int lo = 0, hi = L_;
  while (lo < hi) { int mid = (lo + hi) >> 1; if (m[mid] < w) lo = mid + 1; else hi = mid; }
  int s = lo;
  hi = L_;
  while (lo < hi) { int mid = (lo + hi) >> 1; if (m[mid] < w + 1) lo = mid + 1; else hi = mid; }
  int e = lo;
  int cnt = e - s;
  float inv = 1.0f / (float)(cnt > 1 ? cnt : 1);
  for (int col = threadIdx.x; col < H_; col += blockDim.x) {
    float sum = 0.f;
    for (int r = s; r < e; ++r) sum += tok[(b * L_ + r) * H_ + col];
    wemb[(b * NW_ + w) * H_ + col] = f2bf(sum * inv);
  }
}

// ---------------------------------------------------------------------------
// 3) Input projection GEMM: xg = word_embs @ Wih^T + (bih+bhh)
//    Block tile 128x128, wave tile 16x128. B k-tiles (128x32 bf16, 8KB)
//    double-buffered in LDS via async copies; B frags via ds_load_b128;
//    A frag for k+1 pipelined in registers.
// ---------------------------------------------------------------------------
__global__ void __launch_bounds__(256)
input_proj_kernel(const u16* __restrict__ A, const u16* __restrict__ W,
                  const float* __restrict__ bsum, float* __restrict__ xg) {
  __shared__ u16 bbuf[2][BTILE_];           // 2 x 10KB
  const int tid = threadIdx.x;
  const int wave = tid >> 5, lane = tid & 31;
  const int m0 = blockIdx.x * 128 + wave * 16;
  const int n0 = blockIdx.y * 128;
  const int NK = H_ / 32;                   // 24 k-tiles

  // stage B k-tile kt into bbuf[buf]: 512 x 16B chunks, 2 per thread
  auto prefetch_b = [&](int kt, int buf) {
    int c0 = tid * 2;
#pragma unroll
    for (int u = 0; u < 2; ++u) {
      int c = c0 + u;
      int r = c >> 2, cc = c & 3;           // row 0..127, 16B chunk 0..3
      const u16* g = W + (size_t)(n0 + r) * H_ + kt * 32 + cc * 8;
      unsigned l = (unsigned)(uintptr_t)&bbuf[buf][r * BROW_ + cc * 8];
      async_g2l_b128(l, g);
    }
  };
  auto frag_b_lds = [&](int buf, int j) -> v16bf {
    int r = j * 16 + (lane & 15);
    int kk = (lane >> 4) << 3;
    Frag16 f;
    f.q[0] = *(const v4u*)&bbuf[buf][r * BROW_ + kk];
    f.q[1] = *(const v4u*)&bbuf[buf][r * BROW_ + kk + 16];
    return f.v;
  };

  v8f acc[8] = {};
  prefetch_b(0, 0);
  v16bf a_cur = load_frag_ab(A, H_, m0, 0);
  for (int i = 0; i < NK; ++i) {
    v16bf a_next{};
    if (i + 1 < NK) {
      prefetch_b(i + 1, (i + 1) & 1);       // overlaps with this tile's WMMAs
      a_next = load_frag_ab(A, H_, m0, (i + 1) * 32);
      asm volatile("s_wait_asynccnt 0x2" ::: "memory");   // drain tile i only
    } else {
      asm volatile("s_wait_asynccnt 0x0" ::: "memory");
    }
    __syncthreads();                        // tile i visible to all waves
    v16bf bfr[8];
#pragma unroll
    for (int j = 0; j < 8; ++j) bfr[j] = frag_b_lds(i & 1, j);
#pragma unroll
    for (int j = 0; j < 8; ++j)
      acc[j] = __builtin_amdgcn_wmma_f32_16x16x32_bf16(false, a_cur, false, bfr[j],
                                                       (short)0, acc[j], false, false);
    a_cur = a_next;
    __syncthreads();                        // reads done before buffer reuse
  }

  int mb = m0 + ((lane >> 4) << 3);
#pragma unroll
  for (int j = 0; j < 8; ++j) {
    int n = n0 + j * 16 + (lane & 15);
    float bias = bsum[n];
#pragma unroll
    for (int v = 0; v < 8; ++v) {
      int m = mb + v;
      int t = m & (NW_ - 1);
      int b = m >> 9;                       // NW=512
      xg[(t * B_ + b) * G4_ + n] = acc[j][v] + bias;   // time-major
    }
  }
}

// ---------------------------------------------------------------------------
// 4) Persistent BiLSTM scan. grid=2 (dir), block=512 (16 waves).
//    c in C-fragment registers, h via 24KB LDS bf16; WMMA recurrent matmul;
//    next timestep's 196KB xg slab prefetched (global_prefetch_b8).
// ---------------------------------------------------------------------------
__global__ void __launch_bounds__(512, 1)
lstm_scan_kernel(const float* __restrict__ xg_f, const float* __restrict__ xg_b,
                 const u16* __restrict__ whh_f, const u16* __restrict__ whh_b,
                 u16* __restrict__ hcat) {
  const int dir = blockIdx.x;
  const float* xg = dir ? xg_b : xg_f;
  const u16* whh = dir ? whh_b : whh_f;
  const int dircol = dir * HID_;

  __shared__ u16 h_lds[B_ * HID_];          // 24 KB, bf16 h state
  int tid = threadIdx.x;
  int wave = tid >> 5, lane = tid & 31;
  for (int i = tid; i < B_ * HID_; i += 512) h_lds[i] = 0;
  v8f cfrag[3] = {};
  __syncthreads();

  for (int step = 0; step < NW_; ++step) {
    int t = dir ? (NW_ - 1 - step) : step;
    u16 hreg[3][8];

    if (step + 1 < NW_) {
      int tn = dir ? (t - 1) : (t + 1);
      const float* nb = xg + (size_t)tn * B_ * G4_;
#pragma unroll
      for (int r = 0; r < 3; ++r)
        __builtin_prefetch(nb + (tid * 3 + r) * 32, 0, 0);
    }

    for (int g = 0; g < 3; ++g) {
      int grp = wave * 3 + g;               // 0..47
      int mt = grp / 24;                    // batch tile 0..1
      int jt = grp % 24;                    // hidden tile 0..23
      int jn = jt * 16 + (lane & 15);
      int brow = mt * 16 + ((lane >> 4) << 3);

      v8f acc[4];
#pragma unroll
      for (int gi = 0; gi < 4; ++gi) {
        int n = gi * HID_ + jn;
#pragma unroll
        for (int v = 0; v < 8; ++v)
          acc[gi][v] = xg[(t * B_ + (brow + v)) * G4_ + n];
      }
      for (int k = 0; k < HID_; k += 32) {
        v16bf a = load_frag_ab(h_lds, HID_, mt * 16, k);
        v16bf bfr[4];
#pragma unroll
        for (int gi = 0; gi < 4; ++gi)
          bfr[gi] = load_frag_ab(whh, HID_, gi * HID_ + jt * 16, k);
#pragma unroll
        for (int gi = 0; gi < 4; ++gi)
          acc[gi] = __builtin_amdgcn_wmma_f32_16x16x32_bf16(false, a, false, bfr[gi],
                                                            (short)0, acc[gi], false, false);
      }
#pragma unroll
      for (int v = 0; v < 8; ++v) {
        float ig = sigmoidf_(acc[0][v]);
        float fg = sigmoidf_(acc[1][v]);
        float gg = tanhf(acc[2][v]);
        float og = sigmoidf_(acc[3][v]);
        float c = fg * cfrag[g][v] + ig * gg;
        float h = og * tanhf(c);
        cfrag[g][v] = c;
        u16 hb = f2bf(h);
        hreg[g][v] = hb;
        hcat[((brow + v) * NW_ + t) * H_ + dircol + jn] = hb;
      }
    }
    __syncthreads();
    for (int g = 0; g < 3; ++g) {
      int grp = wave * 3 + g;
      int mt = grp / 24, jt = grp % 24;
      int jn = jt * 16 + (lane & 15);
      int brow = mt * 16 + ((lane >> 4) << 3);
#pragma unroll
      for (int v = 0; v < 8; ++v)
        h_lds[(brow + v) * HID_ + jn] = hreg[g][v];
    }
    __syncthreads();
  }
}

// ---------------------------------------------------------------------------
// 5) Output projection GEMM: seq = hcat @ proj_W^T + proj_b, bf16 out.
//    Same async-LDS staged structure as input projection.
// ---------------------------------------------------------------------------
__global__ void __launch_bounds__(256)
proj_kernel(const u16* __restrict__ A, const u16* __restrict__ W,
            const float* __restrict__ bias, u16* __restrict__ seq) {
  __shared__ u16 bbuf[2][BTILE_];
  const int tid = threadIdx.x;
  const int wave = tid >> 5, lane = tid & 31;
  const int m0 = blockIdx.x * 128 + wave * 16;
  const int n0 = blockIdx.y * 128;
  const int NK = H_ / 32;

  auto prefetch_b = [&](int kt, int buf) {
    int c0 = tid * 2;
#pragma unroll
    for (int u = 0; u < 2; ++u) {
      int c = c0 + u;
      int r = c >> 2, cc = c & 3;
      const u16* g = W + (size_t)(n0 + r) * H_ + kt * 32 + cc * 8;
      unsigned l = (unsigned)(uintptr_t)&bbuf[buf][r * BROW_ + cc * 8];
      async_g2l_b128(l, g);
    }
  };
  auto frag_b_lds = [&](int buf, int j) -> v16bf {
    int r = j * 16 + (lane & 15);
    int kk = (lane >> 4) << 3;
    Frag16 f;
    f.q[0] = *(const v4u*)&bbuf[buf][r * BROW_ + kk];
    f.q[1] = *(const v4u*)&bbuf[buf][r * BROW_ + kk + 16];
    return f.v;
  };

  v8f acc[8] = {};
  prefetch_b(0, 0);
  v16bf a_cur = load_frag_ab(A, H_, m0, 0);
  for (int i = 0; i < NK; ++i) {
    v16bf a_next{};
    if (i + 1 < NK) {
      prefetch_b(i + 1, (i + 1) & 1);
      a_next = load_frag_ab(A, H_, m0, (i + 1) * 32);
      asm volatile("s_wait_asynccnt 0x2" ::: "memory");
    } else {
      asm volatile("s_wait_asynccnt 0x0" ::: "memory");
    }
    __syncthreads();
    v16bf bfr[8];
#pragma unroll
    for (int j = 0; j < 8; ++j) bfr[j] = frag_b_lds(i & 1, j);
#pragma unroll
    for (int j = 0; j < 8; ++j)
      acc[j] = __builtin_amdgcn_wmma_f32_16x16x32_bf16(false, a_cur, false, bfr[j],
                                                       (short)0, acc[j], false, false);
    a_cur = a_next;
    __syncthreads();
  }

  int mb = m0 + ((lane >> 4) << 3);
#pragma unroll
  for (int j = 0; j < 8; ++j) {
    int n = n0 + j * 16 + (lane & 15);
    float bv = bias[n];
#pragma unroll
    for (int v = 0; v < 8; ++v)
      seq[(mb + v) * H_ + n] = f2bf(acc[j][v] + bv);
  }
}

// ---------------------------------------------------------------------------
// 6) Gate MLP + softmax (tiny): one block per batch element.
// ---------------------------------------------------------------------------
__global__ void gate_kernel(const float* __restrict__ tok, const int* __restrict__ lang_ids,
                            const float* __restrict__ ltab,
                            const float* __restrict__ W1, const float* __restrict__ b1,
                            const float* __restrict__ W2, const float* __restrict__ b2,
                            float* __restrict__ gate) {
  int b = blockIdx.x;
  __shared__ float gin[H_ + DL_];
  __shared__ float h1[H_];
  __shared__ float red[E_];
  for (int i = threadIdx.x; i < H_; i += blockDim.x) gin[i] = tok[(size_t)b * L_ * H_ + i];
  int lid = lang_ids[b];
  for (int i = threadIdx.x; i < DL_; i += blockDim.x) gin[H_ + i] = ltab[lid * DL_ + i];
  __syncthreads();
  for (int j = threadIdx.x; j < H_; j += blockDim.x) {
    float s = b1[j];
    for (int k = 0; k < H_ + DL_; ++k) s += gin[k] * W1[j * (H_ + DL_) + k];
    h1[j] = s > 0.f ? s : 0.f;
  }
  __syncthreads();
  if (threadIdx.x < E_) {
    int e = threadIdx.x;
    float s = b2[e];
    for (int k = 0; k < H_; ++k) s += h1[k] * W2[e * H_ + k];
    red[e] = s;
  }
  __syncthreads();
  if (threadIdx.x == 0) {
    float mx = red[0];
    for (int e = 1; e < E_; ++e) mx = red[e] > mx ? red[e] : mx;
    float ex[E_]; float den = 0.f;
    for (int e = 0; e < E_; ++e) { ex[e] = __expf(red[e] - mx); den += ex[e]; }
    for (int e = 0; e < E_; ++e) gate[b * E_ + e] = ex[e] / den;
  }
}

// ---------------------------------------------------------------------------
// 7) Expert heads: fix/dur[b,w] = sum_e gate[b,e] * (seq[b,w]·W[e] + b[e]).
// ---------------------------------------------------------------------------
__global__ void __launch_bounds__(256)
head_kernel(const u16* __restrict__ seq, const float* __restrict__ efW,
            const float* __restrict__ efb, const float* __restrict__ edW,
            const float* __restrict__ edb, const float* __restrict__ gate,
            float* __restrict__ out) {
  int lane = threadIdx.x & 31;
  int wave = threadIdx.x >> 5;
  int row = blockIdx.x * 8 + wave;          // 0..16383  (= b*NW + w)
  int b = row >> 9;
  int w = row & (NW_ - 1);
  float af[E_] = {0.f, 0.f, 0.f, 0.f};
  float ad[E_] = {0.f, 0.f, 0.f, 0.f};
  const u16* srow = seq + (size_t)row * H_;
  for (int k = lane; k < H_; k += 32) {
    float s = bf2f(srow[k]);
#pragma unroll
    for (int e = 0; e < E_; ++e) {
      af[e] += s * efW[e * H_ + k];
      ad[e] += s * edW[e * H_ + k];
    }
  }
#pragma unroll
  for (int e = 0; e < E_; ++e)
    for (int off = 16; off > 0; off >>= 1) {
      af[e] += __shfl_down(af[e], off, 32);
      ad[e] += __shfl_down(ad[e], off, 32);
    }
  if (lane == 0) {
    float fx = 0.f, du = 0.f;
#pragma unroll
    for (int e = 0; e < E_; ++e) {
      float gv = gate[b * E_ + e];
      fx += (af[e] + efb[e]) * gv;
      du += (ad[e] + edb[e]) * gv;
    }
    out[b * NW_ + w] = fx;
    out[B_ * NW_ + b * NW_ + w] = du;
  }
}

// ---------------------------------------------------------------------------
extern "C" void kernel_launch(void* const* d_in, const int* in_sizes, int n_in,
                              void* d_out, int out_size, void* d_ws, size_t ws_size,
                              hipStream_t stream) {
  const float* tok   = (const float*)d_in[0];
  const int*   wmap  = (const int*)d_in[1];
  const int*   lang  = (const int*)d_in[2];
  const float* Wih_f = (const float*)d_in[3];
  const float* Whh_f = (const float*)d_in[4];
  const float* bih_f = (const float*)d_in[5];
  const float* bhh_f = (const float*)d_in[6];
  const float* Wih_b = (const float*)d_in[7];
  const float* Whh_b = (const float*)d_in[8];
  const float* bih_b = (const float*)d_in[9];
  const float* bhh_b = (const float*)d_in[10];
  const float* projW = (const float*)d_in[11];
  const float* projb = (const float*)d_in[12];
  const float* ltab  = (const float*)d_in[13];
  const float* gW1   = (const float*)d_in[14];
  const float* gb1   = (const float*)d_in[15];
  const float* gW2   = (const float*)d_in[16];
  const float* gb2   = (const float*)d_in[17];
  const float* efW   = (const float*)d_in[18];
  const float* efb   = (const float*)d_in[19];
  const float* edW   = (const float*)d_in[20];
  const float* edb   = (const float*)d_in[21];
  float* out = (float*)d_out;

  // workspace carve-out (~280 MB)
  char* ws = (char*)d_ws;
  size_t off = 0;
  auto take = [&](size_t bytes) -> void* {
    void* p = ws + off;
    off = (off + bytes + 255) & ~(size_t)255;
    return p;
  };
  const size_t M = (size_t)B_ * NW_;                 // 16384
  u16*   wemb   = (u16*)  take(M * H_ * 2);          // word_embs bf16
  u16*   wihf16 = (u16*)  take((size_t)G4_ * H_ * 2);
  u16*   wihb16 = (u16*)  take((size_t)G4_ * H_ * 2);
  u16*   whhf16 = (u16*)  take((size_t)G4_ * HID_ * 2);
  u16*   whhb16 = (u16*)  take((size_t)G4_ * HID_ * 2);
  u16*   prjw16 = (u16*)  take((size_t)H_ * H_ * 2);
  float* bsumf  = (float*)take(G4_ * 4);
  float* bsumb  = (float*)take(G4_ * 4);
  float* xgf    = (float*)take((size_t)NW_ * B_ * G4_ * 4);   // time-major
  float* xgb    = (float*)take((size_t)NW_ * B_ * G4_ * 4);
  u16*   hcat   = (u16*)  take(M * H_ * 2);          // [hf|hb] bf16
  u16*   seq16  = (u16*)  take(M * H_ * 2);
  float* gatep  = (float*)take(B_ * E_ * 4);

  // weight conversions / bias folding
  int nwih = G4_ * H_, nwhh = G4_ * HID_, nprj = H_ * H_;
  cvt_kernel<<<(nwih + 255) / 256, 256, 0, stream>>>(Wih_f, wihf16, nwih);
  cvt_kernel<<<(nwih + 255) / 256, 256, 0, stream>>>(Wih_b, wihb16, nwih);
  cvt_kernel<<<(nwhh + 255) / 256, 256, 0, stream>>>(Whh_f, whhf16, nwhh);
  cvt_kernel<<<(nwhh + 255) / 256, 256, 0, stream>>>(Whh_b, whhb16, nwhh);
  cvt_kernel<<<(nprj + 255) / 256, 256, 0, stream>>>(projW, prjw16, nprj);
  bias_sum_kernel<<<(G4_ + 255) / 256, 256, 0, stream>>>(bih_f, bhh_f, bsumf, G4_);
  bias_sum_kernel<<<(G4_ + 255) / 256, 256, 0, stream>>>(bih_b, bhh_b, bsumb, G4_);

  // pooling
  seg_mean_kernel<<<dim3(NW_, B_), 256, 0, stream>>>(tok, wmap, wemb);

  // hoisted input projections (WMMA): M=16384, N=1536, K=768
  input_proj_kernel<<<dim3(128, 12), 256, 0, stream>>>(wemb, wihf16, bsumf, xgf);
  input_proj_kernel<<<dim3(128, 12), 256, 0, stream>>>(wemb, wihb16, bsumb, xgb);

  // recurrent scan, both directions as persistent workgroups
  lstm_scan_kernel<<<2, 512, 0, stream>>>(xgf, xgb, whhf16, whhb16, hcat);

  // output projection (WMMA): M=16384, N=768, K=768
  proj_kernel<<<dim3(128, 6), 256, 0, stream>>>(hcat, prjw16, projb, seq16);

  // gate MLP + expert heads
  gate_kernel<<<B_, 256, 0, stream>>>(tok, lang, ltab, gW1, gb1, gW2, gb2, gatep);
  head_kernel<<<(int)(M / 8), 256, 0, stream>>>(seq16, efW, efb, edW, edb, gatep, out);
}
